// GraphSAGE_78761110274543
// MI455X (gfx1250) — compile-verified, matrix operand order checked by reading, not run on hardware
//
#include <hip/hip_runtime.h>
#include <hip/hip_bf16.h>

// ---------------------------------------------------------------------------
// GraphSAGE forward for MI455X (gfx1250, wave32, WMMA).
//  - Edge aggregation: memory/atomic bound -> coalesced gather + f32 atomics.
//  - Node GEMMs (agg@Wl + x@Wr): v_wmma_f32_16x16x32_f16, f32 accumulate.
//  - Weights repacked once per layer into per-lane WMMA B-fragment layout (f16)
//    so each lane fetches its fragment with one aligned 32B load.
// ---------------------------------------------------------------------------

#define N_NODES  100000
#define N_EDGES  1600000
#define FEAT     128
#define EMB      128
#define DENSE    256
#define N_GRAPHS 512
#define BN_EPS   1e-5f

typedef __attribute__((ext_vector_type(16))) _Float16 v16h;
typedef __attribute__((ext_vector_type(8)))  float    v8f;

// K-index mapping for the 16-bit 16x32 A (and mirrored 32x16 B) fragment:
// per-lane v16h element e, lane-half g (lane/16).  (CDNA5 ISA 7.12.2)
__device__ __forceinline__ int kmap(int e, int g) {
    int v = e >> 1, h = e & 1;
    return (v < 4) ? (8 * g + 2 * v + h)
                   : (16 + 8 * g + 2 * (v - 4) + h);
}

// --------------------------- degree / inv-degree ---------------------------
__global__ void degree_kernel(const int* __restrict__ dst, float* __restrict__ deg) {
    int e = blockIdx.x * 256 + threadIdx.x;
    if (e < N_EDGES) atomicAdd(&deg[dst[e]], 1.0f);
}

__global__ void invdeg_kernel(float* __restrict__ deg) {
    int n = blockIdx.x * 256 + threadIdx.x;
    if (n < N_NODES) deg[n] = 1.0f / fmaxf(deg[n], 1.0f);
}

// --------------------------- edge scatter-add ------------------------------
// 256 threads = 2 edges; lane f = feature -> fully coalesced 512B per edge.
__global__ void scatter_add_kernel(const float* __restrict__ x,
                                   const int* __restrict__ src,
                                   const int* __restrict__ dst,
                                   float* __restrict__ agg) {
    int e = blockIdx.x * 2 + (threadIdx.x >> 7);
    int f = threadIdx.x & 127;
    if (e < N_EDGES) {
        int s = src[e], d = dst[e];
        atomicAdd(&agg[d * EMB + f], x[s * EMB + f]);
    }
}

// --------------------------- weight repack ---------------------------------
// packed element index = ((m*32 + nt*4 + kt)*32 + lane)*16 + e   (f16)
// m: 0=Wl 1=Wr, nt: 16-col tile, kt: 32-K tile.
__global__ void pack_weights_kernel(const float* __restrict__ Wl,
                                    const float* __restrict__ Wr,
                                    _Float16* __restrict__ packed) {
    int tid = blockIdx.x * 256 + threadIdx.x;  // 2*32*32*16 = 32768 total
    if (tid >= 2 * 32 * 32 * 16) return;
    int e    = tid & 15;
    int lane = (tid >> 4) & 31;
    int frag = (tid >> 9) & 31;
    int m    = tid >> 14;
    int nt = frag >> 2, kt = frag & 3;
    int n = nt * 16 + (lane & 15);
    int g = lane >> 4;
    int k = kt * 32 + kmap(e, g);
    const float* W = m ? Wr : Wl;
    packed[tid] = (_Float16)W[k * EMB + n];
}

// --------------------------- SAGE layer (WMMA GEMM) ------------------------
// 128 threads = 4 waves; wave owns 16 nodes; out = relu(bn(agg@Wl + x@Wr + b)).
// out may alias agg: each block only reads/writes its own node rows and the
// agg/x rows are staged into LDS before any store.
__global__ void __launch_bounds__(128)
sage_layer_kernel(const float* __restrict__ agg,
                  const float* __restrict__ xin,
                  const float* __restrict__ invdeg,
                  const _Float16* __restrict__ packedW,
                  const float* __restrict__ bias,
                  const float* __restrict__ gamma,
                  const float* __restrict__ beta,
                  const float* __restrict__ rmean,
                  const float* __restrict__ rvar,
                  float* __restrict__ out) {
    __shared__ _Float16 sAgg[4][16][EMB];
    __shared__ _Float16 sX[4][16][EMB];
    __shared__ float sSc[EMB];
    __shared__ float sSh[EMB];

    const int wave = threadIdx.x >> 5;
    const int lane = threadIdx.x & 31;
    const int node_base = blockIdx.x * 64 + wave * 16;

    // Folded BN scale/shift per output column (blockDim == 128 == EMB).
    {
        int c = threadIdx.x;
        float sc = gamma[c] * rsqrtf(rvar[c] + BN_EPS);
        sSc[c] = sc;
        sSh[c] = (bias[c] - rmean[c]) * sc + beta[c];
    }

    // Stage this wave's 16x128 agg (scaled by inv_deg) and x rows -> f16 LDS.
    #pragma unroll
    for (int i = 0; i < 64; ++i) {
        int idx  = i * 32 + lane;          // 0..2047
        int row  = idx >> 7;
        int col  = idx & 127;
        int node = node_base + row;
        if (node >= N_NODES) node = N_NODES - 1;  // clamp; stores are guarded
        float a = agg[(long)node * EMB + col] * invdeg[node];
        sAgg[wave][row][col] = (_Float16)a;
        sX[wave][row][col]   = (_Float16)xin[(long)node * EMB + col];
    }
    __syncthreads();

    // Build the 4 K-step A fragments for both matrices (held in VGPRs).
    const int row = lane & 15;
    const int g   = lane >> 4;
    v16h aggF[4], xF[4];
    #pragma unroll
    for (int kt = 0; kt < 4; ++kt) {
        #pragma unroll
        for (int e = 0; e < 16; ++e) {
            int k = kt * 32 + kmap(e, g);
            aggF[kt][e] = sAgg[wave][row][k];
            xF[kt][e]   = sX[wave][row][k];
        }
    }

    // 8 column tiles x 4 K-steps x 2 matrices = 64 WMMAs per wave.
    #pragma unroll
    for (int nt = 0; nt < 8; ++nt) {
        v8f acc = {};
        #pragma unroll
        for (int kt = 0; kt < 4; ++kt) {
            const v16h* bl =
                (const v16h*)(packedW + ((((0 * 32 + nt * 4 + kt) * 32) + lane) << 4));
            const v16h* br =
                (const v16h*)(packedW + ((((1 * 32 + nt * 4 + kt) * 32) + lane) << 4));
            v16h bwl = *bl;
            v16h bwr = *br;
            acc = __builtin_amdgcn_wmma_f32_16x16x32_f16(
                false, aggF[kt], false, bwl, (short)0, acc, false, false);
            acc = __builtin_amdgcn_wmma_f32_16x16x32_f16(
                false, xF[kt],   false, bwr, (short)0, acc, false, false);
        }
        // Epilogue: D layout -> lane col = nt*16 + lane%16, row = j + 8*(lane/16).
        int col = nt * 16 + (lane & 15);
        float sc = sSc[col], sh = sSh[col];
        #pragma unroll
        for (int j = 0; j < 8; ++j) {
            int r = j + 8 * g;
            int node = node_base + r;
            if (node < N_NODES) {
                float v = fmaxf(acc[j] * sc + sh, 0.0f);
                out[(long)node * EMB + col] = v;
            }
        }
    }
}

// --------------------------- graph pooling ---------------------------------
// Post-ReLU x >= 0, so uint-bitpattern atomicMax == float max and 0 is the
// correct neutral (matches reference's "finite else 0" for empty graphs).
__global__ void pool_kernel(const float* __restrict__ x,
                            const int* __restrict__ batch,
                            float* __restrict__ gmax,
                            float* __restrict__ gsum,
                            float* __restrict__ gcnt) {
    int node = blockIdx.x * 2 + (threadIdx.x >> 7);
    int f = threadIdx.x & 127;
    if (node < N_NODES) {
        int gidx = batch[node];
        float v = x[(long)node * EMB + f];
        atomicMax((unsigned int*)&gmax[gidx * EMB + f], __float_as_uint(v));
        atomicAdd(&gsum[gidx * EMB + f], v);
        if (f == 0) atomicAdd(&gcnt[gidx], 1.0f);
    }
}

// --------------------------- MLP head --------------------------------------
// One block (256 threads) per graph; all intermediates in LDS.
__global__ void __launch_bounds__(256)
mlp_kernel(const float* __restrict__ gmax, const float* __restrict__ gsum,
           const float* __restrict__ gcnt,
           const float* __restrict__ W1, const float* __restrict__ b1,
           const float* __restrict__ W2, const float* __restrict__ b2,
           const float* __restrict__ W3, const float* __restrict__ b3,
           float* __restrict__ outv) {
    __shared__ float h0[DENSE];
    __shared__ float h1[DENSE];
    __shared__ float red[DENSE];

    int gr = blockIdx.x;
    int t = threadIdx.x;

    float cnt = fmaxf(gcnt[gr], 1.0f);
    h0[t] = (t < EMB) ? gmax[gr * EMB + t]
                      : gsum[gr * EMB + (t - EMB)] / cnt;
    __syncthreads();

    float a = b1[t];
    #pragma unroll 8
    for (int k = 0; k < DENSE; ++k) a = fmaf(h0[k], W1[k * DENSE + t], a);
    h1[t] = fmaxf(a, 0.0f);
    __syncthreads();

    float h2v = 0.0f;
    if (t < DENSE / 2) {
        float b = b2[t];
        #pragma unroll 8
        for (int k = 0; k < DENSE; ++k) b = fmaf(h1[k], W2[k * (DENSE / 2) + t], b);
        h2v = fmaxf(b, 0.0f);
    }
    red[t] = (t < DENSE / 2) ? h2v * W3[t] : 0.0f;
    __syncthreads();
    for (int s = DENSE / 2; s > 0; s >>= 1) {
        if (t < s) red[t] += red[t + s];
        __syncthreads();
    }
    if (t == 0) outv[gr] = red[0] + b3[0];
}

// --------------------------- host launcher ---------------------------------
extern "C" void kernel_launch(void* const* d_in, const int* in_sizes, int n_in,
                              void* d_out, int out_size, void* d_ws, size_t ws_size,
                              hipStream_t stream) {
    const float* x0        = (const float*)d_in[0];
    const int*   eidx      = (const int*)d_in[2];      // [2, N_EDGES]
    const int*   src       = eidx;
    const int*   dst       = eidx + N_EDGES;
    const int*   batch     = (const int*)d_in[3];

    const float* Wl[3] = { (const float*)d_in[4],  (const float*)d_in[7],  (const float*)d_in[10] };
    const float* bl[3] = { (const float*)d_in[5],  (const float*)d_in[8],  (const float*)d_in[11] };
    const float* Wr[3] = { (const float*)d_in[6],  (const float*)d_in[9],  (const float*)d_in[12] };
    const float* gm[3] = { (const float*)d_in[13], (const float*)d_in[17], (const float*)d_in[21] };
    const float* bt[3] = { (const float*)d_in[14], (const float*)d_in[18], (const float*)d_in[22] };
    const float* rm[3] = { (const float*)d_in[15], (const float*)d_in[19], (const float*)d_in[23] };
    const float* rv[3] = { (const float*)d_in[16], (const float*)d_in[20], (const float*)d_in[24] };
    const float* W1 = (const float*)d_in[25]; const float* b1 = (const float*)d_in[26];
    const float* W2 = (const float*)d_in[27]; const float* b2 = (const float*)d_in[28];
    const float* W3 = (const float*)d_in[29]; const float* b3 = (const float*)d_in[30];

    char* ws = (char*)d_ws;
    float*     deg     = (float*)(ws);                                   // 400 KB
    float*     gmaxb   = (float*)(ws + (1u << 20));                      // 256 KB
    float*     gsumb   = (float*)(ws + (1u << 20) + (256u << 10));       // 256 KB
    float*     gcntb   = (float*)(ws + (1u << 20) + (512u << 10));       //   2 KB
    _Float16*  packedW = (_Float16*)(ws + (2u << 20));                   //  64 KB
    float*     buf0    = (float*)(ws + (4u << 20));                      // 51.2 MB
    float*     buf1    = buf0 + (size_t)N_NODES * EMB;                   // 51.2 MB

    const size_t featBytes = (size_t)N_NODES * EMB * sizeof(float);

    // ---- degrees ----
    hipMemsetAsync(deg, 0, N_NODES * sizeof(float), stream);
    degree_kernel<<<(N_EDGES + 255) / 256, 256, 0, stream>>>(dst, deg);
    invdeg_kernel<<<(N_NODES + 255) / 256, 256, 0, stream>>>(deg);

    // ---- 3 SAGE layers (ping-pong buf0/buf1; GEMM out aliases its agg) ----
    const float* xin = x0;
    float* aggb = buf0;
    for (int l = 0; l < 3; ++l) {
        hipMemsetAsync(aggb, 0, featBytes, stream);
        scatter_add_kernel<<<(N_EDGES + 1) / 2, 256, 0, stream>>>(xin, src, dst, aggb);
        pack_weights_kernel<<<128, 256, 0, stream>>>(Wl[l], Wr[l], packedW);
        sage_layer_kernel<<<(N_NODES + 63) / 64, 128, 0, stream>>>(
            aggb, xin, deg, packedW, bl[l], gm[l], bt[l], rm[l], rv[l], aggb);
        xin = aggb;                       // layer output
        aggb = (aggb == buf0) ? buf1 : buf0;
    }

    // ---- pooling ----
    hipMemsetAsync(gmaxb, 0, N_GRAPHS * EMB * sizeof(float), stream);
    hipMemsetAsync(gsumb, 0, N_GRAPHS * EMB * sizeof(float), stream);
    hipMemsetAsync(gcntb, 0, N_GRAPHS * sizeof(float), stream);
    pool_kernel<<<(N_NODES + 1) / 2, 256, 0, stream>>>(xin, batch, gmaxb, gsumb, gcntb);

    // ---- MLP head ----
    mlp_kernel<<<N_GRAPHS, 256, 0, stream>>>(gmaxb, gsumb, gcntb,
                                             W1, b1, W2, b2, W3, b3,
                                             (float*)d_out);
}
